// QuantMultiheadAttention_19052474925379
// MI455X (gfx1250) — matrix operand order repaired and checked by reading, not compiled, not run on hardware
//
#include <hip/hip_runtime.h>
#include <hip/hip_bf16.h>

// ---------------------------------------------------------------------------
// QuantMultiheadAttention forward on gfx1250 (MI455X), bf16 WMMA pipeline.
// B=2, S=2048, E=1024, H=16, D=64.  Compute-bound -> v_wmma_f32_16x16x32_bf16.
// Round 5: fix tensor_load_to_lds arity (this toolchain ships the 6-arg
// builtin: g0 u32x4, g1 i32x8, g2 i32x4, g3 i32x4, i32x8, cpol). Flash K/V
// tiles move via the Tensor Data Mover into double-buffered LDS (TENSORcnt
// pipelined); WMMA fragments read back with aligned ds_load_b128. GEMMs keep
// the 32x64-tile ping-pong register pipeline.
// ---------------------------------------------------------------------------

#define B_ 2
#define S_ 2048
#define E_ 1024
#define H_ 16
#define D_ 64
#define M_ (B_ * S_)        // 4096 rows in the projection GEMMs

typedef __bf16 bf16_t;
typedef __attribute__((ext_vector_type(16))) __bf16 v16bf;
typedef __attribute__((ext_vector_type(8)))  __bf16 v8bf;
typedef __attribute__((ext_vector_type(8)))  float  v8f;
typedef __attribute__((ext_vector_type(4)))  unsigned int v4u32;
typedef __attribute__((ext_vector_type(8)))  int v8i32;
typedef __attribute__((ext_vector_type(4)))  int v4i32;

__device__ __forceinline__ v16bf frag_cat(v8bf lo, v8bf hi) {
    union { v16bf v; v8bf h[2]; } u;
    u.h[0] = lo; u.h[1] = hi;
    return u.v;
}

// A-operand (16xK row slice): per-lane row = lane%16, two 8-elt runs at
// kbase + 8*g and kbase + 16 + 8*g  (ISA 7.12.2, 16-bit A 16x32 layout).
__device__ __forceinline__ v16bf load_a_frag(const bf16_t* __restrict__ p /* row + kbase + 8*g */) {
    v8bf lo = *(const v8bf*)p;
    v8bf hi = *(const v8bf*)(p + 16);
    return frag_cat(lo, hi);
}

// B-operand (Kx16 col slice): per-lane col = lane%16, two 8-elt runs at
// kbase + 8*g and kbase + 16 + 8*g (same k order as the A-operand).
__device__ __forceinline__ v16bf load_b_frag(const bf16_t* __restrict__ p /* col + kbase + 8*g */) {
    v8bf lo = *(const v8bf*)p;
    v8bf hi = *(const v8bf*)(p + 16);
    return frag_cat(lo, hi);
}

// ---------------------------------------------------------------------------
// TDM: 2-D tile load Global -> LDS, bf16 elements (D# per ISA 08 §8.3/8.4).
// tensor: [tensor_d1 rows][tensor_d0 elems], row stride = stride0 elems.
// tile:   [tile_d1 rows][tile_d0 elems] starting at gaddr, packed into LDS.
// ---------------------------------------------------------------------------
__device__ __forceinline__ void tdm_load_2d_bf16(unsigned lds_addr, const void* gaddr,
                                                 unsigned tensor_d0, unsigned tensor_d1,
                                                 unsigned tile_d0, unsigned tile_d1,
                                                 unsigned stride0) {
    const unsigned long long ga = (unsigned long long)gaddr;
    v4u32 g0;
    g0[0] = 1u;                                            // count=1, user descriptor
    g0[1] = lds_addr;                                      // LDS byte address
    g0[2] = (unsigned)(ga & 0xFFFFFFFFu);                  // global_addr[31:0]
    g0[3] = (unsigned)((ga >> 32) & 0x01FFFFFFu) | (2u << 30); // addr[56:32] | type=2
    v8i32 g1;
    g1[0] = (int)(1u << 16);                               // data_size=1 (2 bytes)
    g1[1] = (int)((tensor_d0 & 0xFFFFu) << 16);            // tensor_dim0[15:0]
    g1[2] = (int)((tensor_d0 >> 16) | ((tensor_d1 & 0xFFFFu) << 16));
    g1[3] = (int)((tensor_d1 >> 16) | ((tile_d0 & 0xFFFFu) << 16));
    g1[4] = (int)(tile_d1 & 0xFFFFu);                      // tile_dim1 (tile_dim2=0)
    g1[5] = (int)stride0;                                  // tensor_dim0_stride[31:0]
    g1[6] = 0;
    g1[7] = 0;
    const v4i32 z4 = {0, 0, 0, 0};
    const v8i32 z8 = {0, 0, 0, 0, 0, 0, 0, 0};
    __builtin_amdgcn_tensor_load_to_lds(g0, g1, z4, z4, z8, 0);
}

// ---------------------------------------------------------------------------
// fp32 -> bf16 elementwise convert
// ---------------------------------------------------------------------------
__global__ void cvt_f32_bf16(const float* __restrict__ src, bf16_t* __restrict__ dst, int n) {
    int i = blockIdx.x * blockDim.x + threadIdx.x;
    int stride = gridDim.x * blockDim.x;
    for (; i < n; i += stride) dst[i] = (bf16_t)src[i];
}

// ---------------------------------------------------------------------------
// GEMM: out[m,n] = sum_k X[m,k] * W[n,k]   (X:[M,K] row-major, W:[N,K] row-major)
// One wave per 32x64 output tile; manual 2x-unrolled K loop with ping-pong
// fragment buffers so loads of step i+1 overlap WMMAs of step i.
// MODE: 0=Q (scale+head-split), 1=K (head-split), 2=V (transposed [b,h,d,s]),
// 3=fp32 row-major store.
// ---------------------------------------------------------------------------
template <int MODE>
__global__ __launch_bounds__(32)
void gemm_wmma_bf16(const bf16_t* __restrict__ X, const bf16_t* __restrict__ W,
                    void* __restrict__ out) {
    const int lane = threadIdx.x;
    const int g    = lane >> 4;
    const int i16  = lane & 15;
    const int mbase = blockIdx.x * 32;
    const int nbase = blockIdx.y * 64;

    const bf16_t* arow0 = X + (size_t)(mbase + i16) * E_ + 8 * g;
    const bf16_t* arow1 = arow0 + (size_t)16 * E_;
    const bf16_t* brow[4];
#pragma unroll
    for (int nt = 0; nt < 4; ++nt)
        brow[nt] = W + (size_t)(nbase + nt * 16 + i16) * E_ + 8 * g;

    v8f acc[2][4];
#pragma unroll
    for (int mt = 0; mt < 2; ++mt)
#pragma unroll
        for (int nt = 0; nt < 4; ++nt) acc[mt][nt] = v8f{0};

    // ping buffers: k0 data
    v16bf a0[2], b0[4];
    a0[0] = load_a_frag(arow0);
    a0[1] = load_a_frag(arow1);
#pragma unroll
    for (int nt = 0; nt < 4; ++nt) b0[nt] = load_b_frag(brow[nt]);

    for (int k0 = 0; k0 < E_; k0 += 64) {
        // pong buffers: k0+32 data (loads overlap the WMMAs below)
        v16bf a1[2], b1[4];
        a1[0] = load_a_frag(arow0 + k0 + 32);
        a1[1] = load_a_frag(arow1 + k0 + 32);
#pragma unroll
        for (int nt = 0; nt < 4; ++nt) b1[nt] = load_b_frag(brow[nt] + k0 + 32);

#pragma unroll
        for (int nt = 0; nt < 4; ++nt) {
            acc[0][nt] = __builtin_amdgcn_wmma_f32_16x16x32_bf16(
                false, a0[0], false, b0[nt], (short)0, acc[0][nt], false, false);
            acc[1][nt] = __builtin_amdgcn_wmma_f32_16x16x32_bf16(
                false, a0[1], false, b0[nt], (short)0, acc[1][nt], false, false);
        }

        // refill ping buffers with k0+64 (clamped; redundant reload on last iter)
        const int kn = (k0 + 64 < E_) ? (k0 + 64) : k0;
        a0[0] = load_a_frag(arow0 + kn);
        a0[1] = load_a_frag(arow1 + kn);
#pragma unroll
        for (int nt = 0; nt < 4; ++nt) b0[nt] = load_b_frag(brow[nt] + kn);

#pragma unroll
        for (int nt = 0; nt < 4; ++nt) {
            acc[0][nt] = __builtin_amdgcn_wmma_f32_16x16x32_bf16(
                false, a1[0], false, b1[nt], (short)0, acc[0][nt], false, false);
            acc[1][nt] = __builtin_amdgcn_wmma_f32_16x16x32_bf16(
                false, a1[1], false, b1[nt], (short)0, acc[1][nt], false, false);
        }
    }

#pragma unroll
    for (int mt = 0; mt < 2; ++mt) {
#pragma unroll
        for (int nt = 0; nt < 4; ++nt) {
#pragma unroll
            for (int r = 0; r < 8; ++r) {
                const int m = mbase + mt * 16 + r + 8 * g;
                const int n = nbase + nt * 16 + i16;
                const float v = acc[mt][nt][r];
                if (MODE == 3) {
                    ((float*)out)[(size_t)m * E_ + n] = v;
                } else {
                    const int b = m / S_, s = m % S_;
                    const int h = n / D_, d = n % D_;
                    bf16_t* o = (bf16_t*)out;
                    if (MODE == 0)       // Q, pre-scaled by E^-0.5 = 1/32
                        o[((size_t)(b * H_ + h) * S_ + s) * D_ + d] = (bf16_t)(v * 0.03125f);
                    else if (MODE == 1)  // K
                        o[((size_t)(b * H_ + h) * S_ + s) * D_ + d] = (bf16_t)v;
                    else                 // V transposed: [b,h,d,s]
                        o[((size_t)(b * H_ + h) * D_ + d) * S_ + s] = (bf16_t)v;
                }
            }
        }
    }
}

// ---------------------------------------------------------------------------
// Flash attention: one wave per (head, 16 query rows). K/V tiles are DMA'd
// into double-buffered LDS by the Tensor Data Mover (TENSORcnt pipelined);
// WMMA fragments come from LDS via aligned ds_load_b128. Online softmax runs
// directly on WMMA D-fragments (shfl_xor row reductions; per-lane m/l/corr
// for the 8 rows each lane's accumulators hold). LDS sP stages the P-tile
// D-layout -> A-layout transpose.
// ---------------------------------------------------------------------------
__global__ __launch_bounds__(32)
void flash_attn_wmma(const bf16_t* __restrict__ qh,  // [b,h,s,d] pre-scaled
                     const bf16_t* __restrict__ kh,  // [b,h,s,d]
                     const bf16_t* __restrict__ vT,  // [b,h,d,s]
                     bf16_t* __restrict__ ctx)       // [b,s,e]
{
    __shared__ __align__(16) bf16_t sK[2][32][64];  // K tile: [t][d], 4KB x2
    __shared__ __align__(16) bf16_t sV[2][64][32];  // V tile: [d][t], 4KB x2
    __shared__ __align__(16) bf16_t sP[16][32];     // P tile (A-frag source)

    const int lane = threadIdx.x;
    const int g    = lane >> 4;
    const int i16  = lane & 15;
    const int qt   = blockIdx.x;            // query tile index (16 rows)
    const int bh   = blockIdx.y;            // b*H + h
    const int b    = bh / H_, h = bh % H_;

    const bf16_t* qbase = qh + (size_t)bh * S_ * D_ + (size_t)(qt * 16 + i16) * D_;
    const bf16_t* kbase = kh + (size_t)bh * S_ * D_;
    const bf16_t* vbase = vT + (size_t)bh * D_ * S_;

    const unsigned ldsK0 = (unsigned)(unsigned long long)(void*)&sK[0][0][0];
    const unsigned ldsK1 = (unsigned)(unsigned long long)(void*)&sK[1][0][0];
    const unsigned ldsV0 = (unsigned)(unsigned long long)(void*)&sV[0][0][0];
    const unsigned ldsV1 = (unsigned)(unsigned long long)(void*)&sV[1][0][0];

    v16bf aq[2];
#pragma unroll
    for (int kk = 0; kk < 2; ++kk)
        aq[kk] = load_a_frag(qbase + kk * 32 + 8 * g);

    v8f   o[4] = {v8f{0}, v8f{0}, v8f{0}, v8f{0}};
    float m[8], l[8];
#pragma unroll
    for (int r = 0; r < 8; ++r) { m[r] = -3.0e38f; l[r] = 0.0f; }

    // prologue: DMA tile 0 into buffer 0
    // K tile: rows t0..t0+31 of [S_ x D_]  -> sK[buf][32][64]
    // V tile: cols t0..t0+31 of [D_ x S_]  -> sV[buf][64][32]
    tdm_load_2d_bf16(ldsK0, kbase, D_, S_, D_, 32, D_);
    tdm_load_2d_bf16(ldsV0, vbase, S_, D_, 32, D_, S_);

    for (int t0 = 0; t0 < S_; t0 += 32) {
        const int buf = (t0 >> 5) & 1;

        // DMA next tile into the other buffer, then wait for current tile.
        // TDM ops from one wave complete in order, so tensorcnt<=2 implies
        // the current tile's two loads are resident.
        if (t0 + 32 < S_) {
            tdm_load_2d_bf16(buf ? ldsK0 : ldsK1, kbase + (size_t)(t0 + 32) * D_,
                             D_, S_, D_, 32, D_);
            tdm_load_2d_bf16(buf ? ldsV0 : ldsV1, vbase + (t0 + 32),
                             S_, D_, 32, D_, S_);
            __builtin_amdgcn_s_wait_tensorcnt((short)2);
        } else {
            __builtin_amdgcn_s_wait_tensorcnt((short)0);
        }
        __syncthreads();

        // ---- fragments from LDS ----
        v16bf kb0[2], kb1[2], vb[4];
#pragma unroll
        for (int kk = 0; kk < 2; ++kk) {
            kb0[kk] = load_b_frag(&sK[buf][i16][kk * 32 + 8 * g]);
            kb1[kk] = load_b_frag(&sK[buf][16 + i16][kk * 32 + 8 * g]);
        }
#pragma unroll
        for (int dt = 0; dt < 4; ++dt)
            vb[dt] = load_b_frag(&sV[buf][dt * 16 + i16][8 * g]);

        // ---- scores S[16 x 32] = Q (16x64) @ K_tile^T ----
        v8f c0 = v8f{0}, c1 = v8f{0};
#pragma unroll
        for (int kk = 0; kk < 2; ++kk)
            c0 = __builtin_amdgcn_wmma_f32_16x16x32_bf16(
                false, aq[kk], false, kb0[kk], (short)0, c0, false, false);
#pragma unroll
        for (int kk = 0; kk < 2; ++kk)
            c1 = __builtin_amdgcn_wmma_f32_16x16x32_bf16(
                false, aq[kk], false, kb1[kk], (short)0, c1, false, false);

        // ---- online softmax on fragments; all 32 lanes fully parallel ----
        float corr[8];
#pragma unroll
        for (int r = 0; r < 8; ++r) {
            float tm = fmaxf(c0[r], c1[r]);
#pragma unroll
            for (int mask = 1; mask <= 8; mask <<= 1)
                tm = fmaxf(tm, __shfl_xor(tm, mask, 32));
            const float mn = fmaxf(m[r], tm);
            corr[r] = __expf(m[r] - mn);
            m[r] = mn;
            const float p0 = __expf(c0[r] - mn);
            const float p1 = __expf(c1[r] - mn);
            sP[r + 8 * g][i16]      = (bf16_t)p0;
            sP[r + 8 * g][16 + i16] = (bf16_t)p1;
            float s = p0 + p1;
#pragma unroll
            for (int mask = 1; mask <= 8; mask <<= 1)
                s += __shfl_xor(s, mask, 32);
            l[r] = l[r] * corr[r] + s;
        }
        __syncthreads();

        // ---- P tile as A-fragment from LDS (16B-aligned ds_load_b128) ----
        v8bf plo = *(const v8bf*)&sP[i16][8 * g];
        v8bf phi = *(const v8bf*)&sP[i16][16 + 8 * g];
        v16bf pa = frag_cat(plo, phi);

        // ---- O = O*corr + P @ V_tile  (4 d-tiles of 16) ----
#pragma unroll
        for (int dt = 0; dt < 4; ++dt) {
#pragma unroll
            for (int r = 0; r < 8; ++r) o[dt][r] *= corr[r];
            o[dt] = __builtin_amdgcn_wmma_f32_16x16x32_bf16(
                false, pa, false, vb[dt], (short)0, o[dt], false, false);
        }
        __syncthreads();
    }

    // ---- finalize: O /= l (per-lane stats, no LDS), write ctx bf16 ----
    float inv[8];
#pragma unroll
    for (int r = 0; r < 8; ++r) inv[r] = 1.0f / l[r];

#pragma unroll
    for (int dt = 0; dt < 4; ++dt) {
#pragma unroll
        for (int r = 0; r < 8; ++r) {
            const int s = qt * 16 + r + 8 * g;
            const int e = h * D_ + dt * 16 + i16;
            ctx[((size_t)b * S_ + s) * E_ + e] = (bf16_t)(o[dt][r] * inv[r]);
        }
    }
}

// ---------------------------------------------------------------------------
// Host-side launcher
// ---------------------------------------------------------------------------
extern "C" void kernel_launch(void* const* d_in, const int* in_sizes, int n_in,
                              void* d_out, int out_size, void* d_ws, size_t ws_size,
                              hipStream_t stream) {
    (void)in_sizes; (void)n_in; (void)out_size; (void)ws_size;

    const float* q  = (const float*)d_in[0];
    const float* k  = (const float*)d_in[1];
    const float* v  = (const float*)d_in[2];
    const float* Wq = (const float*)d_in[3];
    const float* Wk = (const float*)d_in[4];
    const float* Wv = (const float*)d_in[5];
    const float* Wo = (const float*)d_in[6];

    const size_t nQKV = (size_t)B_ * S_ * E_;   // 4,194,304
    const size_t nW   = (size_t)E_ * E_;        // 1,048,576

    char* ws = (char*)d_ws;
    size_t off = 0;
    auto take = [&](size_t bytes) -> void* {
        void* p = ws + off;
        off += (bytes + 255) & ~(size_t)255;
        return p;
    };
    bf16_t* qb  = (bf16_t*)take(nQKV * 2);
    bf16_t* kb  = (bf16_t*)take(nQKV * 2);
    bf16_t* vb  = (bf16_t*)take(nQKV * 2);
    bf16_t* wqb = (bf16_t*)take(nW * 2);
    bf16_t* wkb = (bf16_t*)take(nW * 2);
    bf16_t* wvb = (bf16_t*)take(nW * 2);
    bf16_t* wob = (bf16_t*)take(nW * 2);
    bf16_t* qhd = (bf16_t*)take(nQKV * 2);      // [b,h,s,d] scaled
    bf16_t* khd = (bf16_t*)take(nQKV * 2);      // [b,h,s,d]
    bf16_t* vtd = (bf16_t*)take(nQKV * 2);      // [b,h,d,s]
    bf16_t* ctx = (bf16_t*)take(nQKV * 2);      // [b,s,e]

    // 1) convert everything to bf16
    cvt_f32_bf16<<<1024, 256, 0, stream>>>(q,  qb,  (int)nQKV);
    cvt_f32_bf16<<<1024, 256, 0, stream>>>(k,  kb,  (int)nQKV);
    cvt_f32_bf16<<<1024, 256, 0, stream>>>(v,  vb,  (int)nQKV);
    cvt_f32_bf16<<<512,  256, 0, stream>>>(Wq, wqb, (int)nW);
    cvt_f32_bf16<<<512,  256, 0, stream>>>(Wk, wkb, (int)nW);
    cvt_f32_bf16<<<512,  256, 0, stream>>>(Wv, wvb, (int)nW);
    cvt_f32_bf16<<<512,  256, 0, stream>>>(Wo, wob, (int)nW);

    // 2) Q/K/V projections (WMMA), head-split layouts; Q pre-scaled; V transposed
    dim3 gproj(M_ / 32, E_ / 64);   // 128 x 16 tiles, one wave32 each
    gemm_wmma_bf16<0><<<gproj, 32, 0, stream>>>(qb, wqb, qhd);
    gemm_wmma_bf16<1><<<gproj, 32, 0, stream>>>(kb, wkb, khd);
    gemm_wmma_bf16<2><<<gproj, 32, 0, stream>>>(vb, wvb, vtd);

    // 3) attention: TDM-fed, WMMA-based flash kernel
    dim3 gattn(S_ / 16, B_ * H_);   // 128 x 32 waves
    flash_attn_wmma<<<gattn, 32, 0, stream>>>(qhd, khd, vtd, ctx);

    // 4) output projection -> fp32 d_out
    gemm_wmma_bf16<3><<<gproj, 32, 0, stream>>>(ctx, wob, d_out);
}
